// SetAbstraction_9998683865703
// MI455X (gfx1250) — matrix operand order repaired and checked by reading, not compiled, not run on hardware
//
#include <hip/hip_runtime.h>
#include <hip/hip_bf16.h>

#define NPTS 100000
#define KC   1024      // centroids AND max neighbors
#define CIN  64
#define RSQ  0.04f     // R*R
#define D1o  64
#define D2o  64
#define D3o  128
#define KP   96        // layer-1 K dim padded: [rel0,rel1,rel2, 0, f0..f63, 0*28]

typedef __attribute__((ext_vector_type(16))) __bf16 v16bf;
typedef __attribute__((ext_vector_type(8)))  float  v8f;

union FragBF { v16bf v; uint4 q[2]; };

// round-to-nearest-even f32 -> bf16
__device__ __forceinline__ unsigned short f2bf(float f) {
    unsigned int x = __float_as_uint(f);
    x += 0x7FFFu + ((x >> 16) & 1u);
    return (unsigned short)(x >> 16);
}

// A fragment (16x32 bf16): lane l<16 row M=l holds K {k0..k0+7, k0+16..k0+23},
// lanes 16..31 same rows, K shifted by +8.  Two ds_load_b128 per fragment.
__device__ __forceinline__ v16bf ldA(const unsigned short* X, int stride, int row, int k0, int lh) {
    FragBF f;
    const unsigned short* p = X + row * stride + k0 + lh * 8;
    f.q[0] = *(const uint4*)(p);
    f.q[1] = *(const uint4*)(p + 16);
    return f.v;
}

// B fragment (32x16 bf16) from column-major W^T[n][k]: lane l<16 col N=n holds K
// {k0..k0+7, k0+16..k0+23}; lanes 16..31 K shifted by +8.  Two ds_load_b128.
__device__ __forceinline__ v16bf ldB(const unsigned short* WT, int kstride, int n, int k0, int lh) {
    FragBF f;
    const unsigned short* p = WT + n * kstride + k0 + lh * 8;
    f.q[0] = *(const uint4*)(p);
    f.q[1] = *(const uint4*)(p + 16);
    return f.v;
}

// ---------------- Kernel 1: ball query (order-preserving first-K) ----------------
__global__ __launch_bounds__(1024)
void ball_query_kernel(const float* __restrict__ coords,
                       const int* __restrict__ centroid_idx,
                       int* __restrict__ idxbuf, int* __restrict__ cntbuf) {
    const int k    = blockIdx.x;
    const int tid  = threadIdx.x;
    const int lane = tid & 31;
    const int w    = tid >> 5;

    __shared__ int   s_wcnt[32];
    __shared__ int   s_base;
    __shared__ float s_c[3];
    if (tid == 0) {
        int ci = centroid_idx[k];
        s_c[0] = coords[ci * 3 + 0];
        s_c[1] = coords[ci * 3 + 1];
        s_c[2] = coords[ci * 3 + 2];
        s_base = 0;
    }
    __syncthreads();
    const float cx = s_c[0], cy = s_c[1], cz = s_c[2];

    for (int base = 0; base < NPTS; base += 1024) {
        int i = base + tid;
        bool m = false;
        if (i < NPTS) {
            float dx = coords[i * 3 + 0] - cx;
            float dy = coords[i * 3 + 1] - cy;
            float dz = coords[i * 3 + 2] - cz;
            m = (dx * dx + dy * dy + dz * dz) <= RSQ;
        }
        unsigned bal = (unsigned)__ballot(m);         // wave32: low 32 bits
        if (lane == 0) s_wcnt[w] = __popc(bal);
        __syncthreads();
        if (tid == 0) {                               // exclusive scan over 32 waves
            int acc = s_base;
            #pragma unroll
            for (int j = 0; j < 32; ++j) { int c = s_wcnt[j]; s_wcnt[j] = acc; acc += c; }
            s_base = acc;
        }
        __syncthreads();
        int pos = s_wcnt[w] + __popc(bal & ((1u << lane) - 1u));
        if (m && pos < KC) idxbuf[k * KC + pos] = i;
        __syncthreads();                              // protect s_wcnt/s_base for next iter
        if (s_base >= KC) break;                      // uniform early exit
    }
    if (tid == 0) cntbuf[k] = s_base < KC ? s_base : KC;
}

// -------- Kernel 2: fused gather + bf16-WMMA MLP (67->64->64->128) + maxpool ------
__global__ __launch_bounds__(256)
void sa_mlp_kernel(const float* __restrict__ coords,
                   const float* __restrict__ features,
                   const int* __restrict__ centroid_idx,
                   const float* __restrict__ W1, const float* __restrict__ B1,
                   const float* __restrict__ W2, const float* __restrict__ B2,
                   const float* __restrict__ W3, const float* __restrict__ B3,
                   const int* __restrict__ idxbuf, const int* __restrict__ cntbuf,
                   float* __restrict__ out) {
    const int k   = blockIdx.x;
    const int tid = threadIdx.x;
    const int lane = tid & 31, w = tid >> 5;
    const int l15 = lane & 15, lh = lane >> 4;

    __shared__ __align__(16) unsigned short sX0[128 * KP];   // 24 KB activations (bf16)
    __shared__ __align__(16) unsigned short sX1[128 * D1o];  // 16 KB
    __shared__ __align__(16) unsigned short sX2[128 * D2o];  // 16 KB
    __shared__ __align__(16) unsigned short sW1[D1o * KP];   // 12 KB  W1^T [64][96]
    __shared__ __align__(16) unsigned short sW2[D2o * D1o];  //  8 KB  W2^T [64][64]
    __shared__ __align__(16) unsigned short sW3[D3o * D2o];  // 16 KB  W3^T [128][64]
    __shared__ float sB1[D1o], sB2[D2o], sB3[D3o];
    __shared__ float sMax[D3o];
    __shared__ float sWaveMax[8 * D3o];
    __shared__ int   sCnt;
    __shared__ float sCent[3];

    // ---- preload weights (transposed, bf16, zero-padded), biases, state ----
    for (int t = tid; t < D1o * KP; t += 256) {     // W1^T: x cols are [rel*3, pad, feat*64, pad]
        int n = t / KP, kk = t % KP;
        float v = 0.0f;
        if (kk < 3)                 v = W1[kk * D1o + n];
        else if (kk >= 4 && kk < 4 + CIN) v = W1[(kk - 1) * D1o + n];
        sW1[n * KP + kk] = f2bf(v);
    }
    for (int t = tid; t < D2o * D1o; t += 256) {
        int n = t / D1o, kk = t % D1o;
        sW2[n * D1o + kk] = f2bf(W2[kk * D2o + n]);
    }
    for (int t = tid; t < D3o * D2o; t += 256) {
        int n = t / D2o, kk = t % D2o;
        sW3[n * D2o + kk] = f2bf(W3[kk * D3o + n]);
    }
    for (int t = tid; t < D1o; t += 256) sB1[t] = B1[t];
    for (int t = tid; t < D2o; t += 256) sB2[t] = B2[t];
    for (int t = tid; t < D3o; t += 256) sB3[t] = B3[t];
    if (tid < D3o) sMax[tid] = -__builtin_inff();
    // zero the static pad columns of X0 (col 3 and cols 68..95) once
    for (int t = tid; t < 128 * 29; t += 256) {
        int r = t / 29, c = t % 29;
        sX0[r * KP + (c == 0 ? 3 : 67 + c)] = 0;
    }
    if (tid == 0) sCnt = cntbuf[k];
    if (tid < 3) {
        int ci = centroid_idx[k];
        sCent[tid] = coords[ci * 3 + tid];
    }
    __syncthreads();

    const int   cnt = sCnt;
    const float c0 = sCent[0], c1 = sCent[1], c2 = sCent[2];
    const int Mbase = w * 16;

    for (int t = 0; t < 8; ++t) {                    // 8 row-tiles of 128 neighbors
        const int tbase = t * 128;

        // ---- gather: 2 threads per row; bf16 into LDS ----
        {
            const int row = tid >> 1, h = tid & 1;
            const int slot = tbase + row;
            int j = 0;
            if (slot < cnt) j = idxbuf[k * KC + slot];
            if (h == 0) {
                sX0[row * KP + 0] = f2bf(coords[j * 3 + 0] - c0);
                sX0[row * KP + 1] = f2bf(coords[j * 3 + 1] - c1);
                sX0[row * KP + 2] = f2bf(coords[j * 3 + 2] - c2);
            }
            const float4* fp = (const float4*)(features + (size_t)j * CIN) + h * 8;
            unsigned short* xp = sX0 + row * KP + 4 + h * 32;
            #pragma unroll
            for (int q = 0; q < 8; ++q) {
                float4 f = fp[q];
                ushort4 s4 = { f2bf(f.x), f2bf(f.y), f2bf(f.z), f2bf(f.w) };
                *(ushort4*)(xp + q * 4) = s4;
            }
        }
        __syncthreads();

        // ---- layer 1: [128,96]bf16 @ [96,64] -> relu -> sX1 ----
        #pragma unroll
        for (int nt = 0; nt < 4; ++nt) {
            v8f c = {};
            #pragma unroll
            for (int k0 = 0; k0 < KP; k0 += 32) {
                v16bf a = ldA(sX0, KP, Mbase + l15, k0, lh);
                v16bf b = ldB(sW1, KP, nt * 16 + l15, k0, lh);
                c = __builtin_amdgcn_wmma_f32_16x16x32_bf16(false, a, false, b,
                                                            (short)0, c, false, false);
            }
            const int col = nt * 16 + l15;
            const float bias = sB1[col];
            #pragma unroll
            for (int r = 0; r < 8; ++r) {
                float v = c[r] + bias;
                v = v > 0.0f ? v : 0.0f;
                sX1[(Mbase + r + lh * 8) * D1o + col] = f2bf(v);
            }
        }
        __syncthreads();

        // ---- layer 2: [128,64] @ [64,64] -> relu -> sX2 ----
        #pragma unroll
        for (int nt = 0; nt < 4; ++nt) {
            v8f c = {};
            #pragma unroll
            for (int k0 = 0; k0 < D1o; k0 += 32) {
                v16bf a = ldA(sX1, D1o, Mbase + l15, k0, lh);
                v16bf b = ldB(sW2, D1o, nt * 16 + l15, k0, lh);
                c = __builtin_amdgcn_wmma_f32_16x16x32_bf16(false, a, false, b,
                                                            (short)0, c, false, false);
            }
            const int col = nt * 16 + l15;
            const float bias = sB2[col];
            #pragma unroll
            for (int r = 0; r < 8; ++r) {
                float v = c[r] + bias;
                v = v > 0.0f ? v : 0.0f;
                sX2[(Mbase + r + lh * 8) * D2o + col] = f2bf(v);
            }
        }
        __syncthreads();

        // ---- layer 3: [128,64] @ [64,128] -> relu -> mask -> wave max ----
        #pragma unroll
        for (int nt = 0; nt < 8; ++nt) {
            v8f c = {};
            #pragma unroll
            for (int k0 = 0; k0 < D2o; k0 += 32) {
                v16bf a = ldA(sX2, D2o, Mbase + l15, k0, lh);
                v16bf b = ldB(sW3, D2o, nt * 16 + l15, k0, lh);
                c = __builtin_amdgcn_wmma_f32_16x16x32_bf16(false, a, false, b,
                                                            (short)0, c, false, false);
            }
            const int col = nt * 16 + l15;
            const float bias = sB3[col];
            float m = -__builtin_inff();
            #pragma unroll
            for (int r = 0; r < 8; ++r) {
                int g = tbase + Mbase + r + lh * 8;
                float v = c[r] + bias;
                v = v > 0.0f ? v : 0.0f;                 // relu then mask (matches ref)
                v = (g < cnt) ? v : -__builtin_inff();
                m = m > v ? m : v;
            }
            float o = __shfl_xor(m, 16);                 // combine lane halves (same col)
            m = m > o ? m : o;
            if (lh == 0) sWaveMax[w * D3o + col] = m;
        }
        __syncthreads();
        if (tid < D3o) {                                 // fold 8 wave maxes into running max
            float v = sMax[tid];
            #pragma unroll
            for (int ww = 0; ww < 8; ++ww) {
                float x = sWaveMax[ww * D3o + tid];
                v = v > x ? v : x;
            }
            sMax[tid] = v;
        }
        __syncthreads();
    }

    // outputs: cent [K,3] first, then new_features [K,128]
    if (tid < 3)   out[k * 3 + tid] = sCent[tid];
    if (tid < D3o) out[3 * KC + k * D3o + tid] = sMax[tid];
}

extern "C" void kernel_launch(void* const* d_in, const int* in_sizes, int n_in,
                              void* d_out, int out_size, void* d_ws, size_t ws_size,
                              hipStream_t stream) {
    (void)in_sizes; (void)n_in; (void)out_size; (void)ws_size;
    const float* coords       = (const float*)d_in[0];
    const float* features     = (const float*)d_in[1];
    const int*   centroid_idx = (const int*)d_in[2];
    const float* W1 = (const float*)d_in[3];
    const float* B1 = (const float*)d_in[4];
    const float* W2 = (const float*)d_in[5];
    const float* B2 = (const float*)d_in[6];
    const float* W3 = (const float*)d_in[7];
    const float* B3 = (const float*)d_in[8];

    int* idxbuf = (int*)d_ws;                 // [K][K] first-K in-radius indices
    int* cntbuf = idxbuf + (size_t)KC * KC;   // [K] valid counts

    ball_query_kernel<<<KC, 1024, 0, stream>>>(coords, centroid_idx, idxbuf, cntbuf);
    sa_mlp_kernel<<<KC, 256, 0, stream>>>(coords, features, centroid_idx,
                                          W1, B1, W2, B2, W3, B3,
                                          idxbuf, cntbuf, (float*)d_out);
}
